// MyLSTMCell_35725537968252
// MI455X (gfx1250) — matrix operand order repaired and checked by reading, not compiled, not run on hardware
//
#include <hip/hip_runtime.h>
#include <hip/hip_bf16.h>

typedef __attribute__((ext_vector_type(16))) __bf16 v16bf;
typedef __attribute__((ext_vector_type(8)))  __bf16 v8bf;
typedef __attribute__((ext_vector_type(8)))  float  v8f;

#define B_ROWS 4096
#define K_IN   1024
#define H_DIM  1024

#define BM 64            // rows per workgroup
#define BN 64            // h-columns per workgroup (x4 gates)
#define KB 32            // k per stage (one WMMA K-step)
#define KPAD 40          // KB + 8 bf16 (16B) padding vs bank conflicts
#define NSTAGES ((K_IN + H_DIM) / KB)   // 64 stages over concatenated K

__device__ __forceinline__ float sigmoid_f(float x) {
    return 1.0f / (1.0f + __expf(-x));
}

union Frag16 { v16bf v; v8bf h[2]; };

__global__ __launch_bounds__(256)
void lstm_cell_wmma_bf16(
    const float* __restrict__ x,      // (B, K_IN)
    const float* __restrict__ hprev,  // (B, H)
    const float* __restrict__ cprev,  // (B, H)
    const float* __restrict__ Wxi, const float* __restrict__ Wxf,
    const float* __restrict__ Wxg, const float* __restrict__ Wxo,
    const float* __restrict__ Whi, const float* __restrict__ Whf,
    const float* __restrict__ Whg, const float* __restrict__ Who,
    const float* __restrict__ bxi, const float* __restrict__ bxf,
    const float* __restrict__ bxg, const float* __restrict__ bxo,
    const float* __restrict__ bhi, const float* __restrict__ bhf,
    const float* __restrict__ bhg, const float* __restrict__ bho,
    float* __restrict__ out_h, float* __restrict__ out_c)
{
    __shared__ __align__(32) __bf16 As[2][BM][KPAD];        //  10 KB
    __shared__ __align__(32) __bf16 Ws[2][4][BN][KPAD];     //  40 KB

    const int tid  = threadIdx.x;
    const int lane = tid & 31;
    const int wave = tid >> 5;
    const int wm   = wave & 1;   // 0..1 : 32-row strip (2 x 16-row subtiles)
    const int wn   = wave >> 1;  // 0..3 : 16-col strip

    const int m0 = blockIdx.x * BM;
    const int n0 = blockIdx.y * BN;

    // ---- accumulators: 4 gates x 2 m-subtiles of 16x16 f32 ----
    v8f acc[4][2];
    const v8f vzero = {0.f,0.f,0.f,0.f,0.f,0.f,0.f,0.f};
    #pragma unroll
    for (int g = 0; g < 4; ++g)
        #pragma unroll
        for (int s = 0; s < 2; ++s)
            acc[g][s] = vzero;

    // ---- stage-loader thread mapping ----
    const int a_row  = tid >> 2;   // 0..63  (A tile row)
    const int a_quad = tid & 3;    // 0..3   (8 floats each)
    const int w_row  = tid & 63;   // 0..63  (W tile n-row)

    // gate index is wave-uniform (tid>>6): force it scalar so the weight
    // pointer select stays in SGPRs (no scratch-spilled pointer array).
    const int gate_s = __builtin_amdgcn_readfirstlane(tid >> 6);
    const float* Wxg_s = (gate_s == 0) ? Wxi : (gate_s == 1) ? Wxf
                       : (gate_s == 2) ? Wxg : Wxo;
    const float* Whg_s = (gate_s == 0) ? Whi : (gate_s == 1) ? Whf
                       : (gate_s == 2) ? Whg : Who;

    // Per-thread base pointers, H-phase biased by -K_IN so `base + kb` works
    // for the whole concatenated-K loop with no per-stage index math.
    const float* aX = x     + (size_t)(m0 + a_row) * K_IN;
    const float* aH = hprev + (size_t)(m0 + a_row) * H_DIM - K_IN;
    const float* wX = Wxg_s + (size_t)(n0 + w_row) * K_IN;
    const float* wH = Whg_s + (size_t)(n0 + w_row) * K_IN - K_IN;

    // global f32 -> bf16 -> LDS (NT layout: both operands contiguous in k)
    auto load_stage = [&](int s, int buf) {
        const int kb = s * KB;
        {   // A tile: rows of [X | h_prev]
            const float* src = ((kb < K_IN) ? aX : aH) + kb;
            const float4 a0 = ((const float4*)src)[a_quad * 2 + 0];
            const float4 a1 = ((const float4*)src)[a_quad * 2 + 1];
            v8bf p;
            p[0]=(__bf16)a0.x; p[1]=(__bf16)a0.y; p[2]=(__bf16)a0.z; p[3]=(__bf16)a0.w;
            p[4]=(__bf16)a1.x; p[5]=(__bf16)a1.y; p[6]=(__bf16)a1.z; p[7]=(__bf16)a1.w;
            *(v8bf*)&As[buf][a_row][a_quad * 8] = p;
        }
        {   // W tile: one 32-float row per thread for this wave's gate
            const float* wsrc = ((kb < K_IN) ? wX : wH) + kb;
            #pragma unroll
            for (int q = 0; q < 4; ++q) {
                const float4 w0 = ((const float4*)wsrc)[q * 2 + 0];
                const float4 w1 = ((const float4*)wsrc)[q * 2 + 1];
                v8bf p;
                p[0]=(__bf16)w0.x; p[1]=(__bf16)w0.y; p[2]=(__bf16)w0.z; p[3]=(__bf16)w0.w;
                p[4]=(__bf16)w1.x; p[5]=(__bf16)w1.y; p[6]=(__bf16)w1.z; p[7]=(__bf16)w1.w;
                *(v8bf*)&Ws[buf][gate_s][w_row][q * 8] = p;
            }
        }
    };

    // ---- one k-stage: 2 A frags + 4 B frags -> 8 WMMAs (B reused x2) ----
    const int half8  = (lane >> 4) * 8;   // A frag: lane-half selects k sub-block
    const int b_lrow = wn * 16 + (lane & 15);
    auto compute_stage = [&](int buf) {
        Frag16 a[2];
        #pragma unroll
        for (int ms = 0; ms < 2; ++ms) {
            const __bf16* Ap = &As[buf][wm * 32 + ms * 16 + (lane & 15)][0];
            a[ms].h[0] = *(const v8bf*)(Ap + half8);
            a[ms].h[1] = *(const v8bf*)(Ap + 16 + half8);
        }
        #pragma unroll
        for (int g = 0; g < 4; ++g) {
            Frag16 b;
            const __bf16* Bp = &Ws[buf][g][b_lrow][(lane >> 4) * 8];
            b.h[0] = *(const v8bf*)(Bp + 0);
            b.h[1] = *(const v8bf*)(Bp + 16);
            #pragma unroll
            for (int ms = 0; ms < 2; ++ms) {
                acc[g][ms] = __builtin_amdgcn_wmma_f32_16x16x32_bf16(
                    false, a[ms].v, false, b.v, (short)0, acc[g][ms], false, false);
            }
        }
    };

    load_stage(0, 0);
    __syncthreads();
    for (int s = 0; s < NSTAGES; ++s) {
        if (s + 1 < NSTAGES) load_stage(s + 1, (s + 1) & 1);
        if (s + 2 < NSTAGES) {   // gfx1250 global_prefetch two stages ahead
            const int kb2 = (s + 2) * KB;
            __builtin_prefetch((((kb2 < K_IN) ? aX : aH) + kb2) + a_quad * 8, 0, 3);
            __builtin_prefetch((((kb2 < K_IN) ? wX : wH) + kb2), 0, 3);
        }
        compute_stage(s & 1);
        __syncthreads();
    }

    // ---- fused LSTM epilogue ----
    // C layout: VGPR r, lane L -> M = 8*(L/16)+r, N = L%16
    const int h  = n0 + wn * 16 + (lane & 15);
    const float bi = bxi[h] + bhi[h];
    const float bf = bxf[h] + bhf[h];
    const float bg = bxg[h] + bhg[h];
    const float bo = bxo[h] + bho[h];
    #pragma unroll
    for (int ms = 0; ms < 2; ++ms) {
        const int mbase = m0 + wm * 32 + ms * 16 + (lane >> 4) * 8;
        #pragma unroll
        for (int r = 0; r < 8; ++r) {
            const int m = mbase + r;
            const float gi = sigmoid_f(acc[0][ms][r] + bi);
            const float gf = sigmoid_f(acc[1][ms][r] + bf);
            const float gg = tanhf(acc[2][ms][r] + bg);
            const float go = sigmoid_f(acc[3][ms][r] + bo);
            const float c  = gf * cprev[(size_t)m * H_DIM + h] + gi * gg;
            out_h[(size_t)m * H_DIM + h] = go * tanhf(c);
            out_c[(size_t)m * H_DIM + h] = c;
        }
    }
}

extern "C" void kernel_launch(void* const* d_in, const int* in_sizes, int n_in,
                              void* d_out, int out_size, void* d_ws, size_t ws_size,
                              hipStream_t stream) {
    (void)in_sizes; (void)n_in; (void)out_size; (void)d_ws; (void)ws_size;
    // setup_inputs() dict order:
    // 0:input_ 1:prev_h 2:prev_c then (W,b) pairs: ii,hi,if_,hf,ig,hg,io,ho
    const float* x     = (const float*)d_in[0];
    const float* hprev = (const float*)d_in[1];
    const float* cprev = (const float*)d_in[2];
    const float* W_ii = (const float*)d_in[3];  const float* b_ii = (const float*)d_in[4];
    const float* W_hi = (const float*)d_in[5];  const float* b_hi = (const float*)d_in[6];
    const float* W_if = (const float*)d_in[7];  const float* b_if = (const float*)d_in[8];
    const float* W_hf = (const float*)d_in[9];  const float* b_hf = (const float*)d_in[10];
    const float* W_ig = (const float*)d_in[11]; const float* b_ig = (const float*)d_in[12];
    const float* W_hg = (const float*)d_in[13]; const float* b_hg = (const float*)d_in[14];
    const float* W_io = (const float*)d_in[15]; const float* b_io = (const float*)d_in[16];
    const float* W_ho = (const float*)d_in[17]; const float* b_ho = (const float*)d_in[18];

    float* out_h = (float*)d_out;                         // tuple (h, c) flat
    float* out_c = out_h + (size_t)B_ROWS * H_DIM;

    dim3 grid(B_ROWS / BM, H_DIM / BN);   // 64 x 16
    lstm_cell_wmma_bf16<<<grid, 256, 0, stream>>>(
        x, hprev, cprev,
        W_ii, W_if, W_ig, W_io,
        W_hi, W_hf, W_hg, W_ho,
        b_ii, b_if, b_ig, b_io,
        b_hi, b_hf, b_hg, b_ho,
        out_h, out_c);
}
